// AttnGCN_7687991460229
// MI455X (gfx1250) — matrix operand order repaired and checked by reading, not compiled, not run on hardware
//
#include <hip/hip_runtime.h>
#include <hip/hip_bf16.h>
#include <math.h>

#define N_NODES 100000
#define N_EDGES 800000
#define ET      (N_EDGES + N_NODES)   // 900000 with self loops
#define HEADS   2
#define CDIM    128
#define HC      (HEADS * CDIM)        // 256
#define INF     5
#define OUTF    5
#define ROWP    136                   // 128 + 8 ushort pad -> 272B row stride, staggers LDS banks

typedef __bf16 v16bf __attribute__((ext_vector_type(16)));
typedef float  v8f   __attribute__((ext_vector_type(8)));
typedef unsigned uint4v __attribute__((ext_vector_type(4)));
typedef int v4i_vs __attribute__((vector_size(16)));   // matches async-LDS builtin param type

union Frag16 { v16bf v; uint4v q[2]; };

#if __has_builtin(__builtin_amdgcn_global_load_async_to_lds_b128)
#define USE_ASYNC_LDS 1
#endif

// ---------- helpers ----------
__device__ __forceinline__ unsigned enc_ord(float f) {
    unsigned b = __float_as_uint(f);
    return (b & 0x80000000u) ? ~b : (b | 0x80000000u);
}
__device__ __forceinline__ float dec_ord(unsigned u) {
    unsigned b = (u & 0x80000000u) ? (u & 0x7FFFFFFFu) : ~u;
    return __uint_as_float(b);
}
__device__ __forceinline__ unsigned short f2bf(float f) {  // RNE
    unsigned b = __float_as_uint(f);
    unsigned r = b + 0x7FFFu + ((b >> 16) & 1u);
    return (unsigned short)(r >> 16);
}
__device__ __forceinline__ float lrelu(float v, float s) { return v > 0.f ? v : s * v; }

// 16B global -> LDS copy: async DMA path on gfx1250, VGPR staging fallback
__device__ __forceinline__ void cp16(const unsigned short* g, unsigned short* l) {
#ifdef USE_ASYNC_LDS
    __builtin_amdgcn_global_load_async_to_lds_b128(
        (__attribute__((address_space(1))) v4i_vs*)g,
        (__attribute__((address_space(3))) v4i_vs*)l, 0, 0);
#else
    *(uint4v*)l = *(const uint4v*)g;
#endif
}
__device__ __forceinline__ void async_wait0() {
#ifdef USE_ASYNC_LDS
#if __has_builtin(__builtin_amdgcn_s_wait_asynccnt)
    __builtin_amdgcn_s_wait_asynccnt(0);
#else
    asm volatile("s_wait_asynccnt 0" ::: "memory");
#endif
#endif
}

// ---------- init ----------
__global__ void fill_u32(unsigned* p, unsigned v, int n) {
    int t = blockIdx.x * blockDim.x + threadIdx.x;
    if (t < n) p[t] = v;
}

// ---------- edge-weight mean ----------
__global__ void ew_sum(const float* __restrict__ ew, float* out) {
    __shared__ float s[256];
    float acc = 0.f;
    for (int i = blockIdx.x * blockDim.x + threadIdx.x; i < N_EDGES; i += gridDim.x * blockDim.x)
        acc += ew[i];
    s[threadIdx.x] = acc; __syncthreads();
    for (int o = 128; o > 0; o >>= 1) {
        if (threadIdx.x < o) s[threadIdx.x] += s[threadIdx.x + o];
        __syncthreads();
    }
    if (threadIdx.x == 0) atomicAdd(out, s[0]);
}

// ---------- in-degree (shared by both layers) ----------
__global__ void deg_count(const int* __restrict__ ei, float* deg) {
    int e = blockIdx.x * blockDim.x + threadIdx.x;
    if (e >= ET) return;
    int t = (e < N_EDGES) ? ei[N_EDGES + e] : (e - N_EDGES);
    atomicAdd(&deg[t], 1.0f);
}

// ---------- layer-1 projection (K=5: VALU is the right tool) ----------
__global__ void proj1(const float* __restrict__ h,
                      const float* __restrict__ Wl, const float* __restrict__ bl,
                      const float* __restrict__ Wr, const float* __restrict__ br,
                      float* __restrict__ xl, float* __restrict__ xr) {
    int t = blockIdx.x * blockDim.x + threadIdx.x;
    if (t >= N_NODES * HC) return;
    int n = t / HC, c = t % HC;
    const float* hr = h + (size_t)n * INF;
    float al = bl[c], ar = br[c];
#pragma unroll
    for (int k = 0; k < INF; k++) {
        float hv = hr[k];
        al = fmaf(hv, Wl[k * HC + c], al);
        ar = fmaf(hv, Wr[k * HC + c], ar);
    }
    xl[t] = al; xr[t] = ar;
}

// ---------- attention logits: one wave per (edge, head) ----------
__global__ void edge_alpha(const int* __restrict__ ei, const float* __restrict__ ew,
                           const float* __restrict__ ewsum,
                           const float* __restrict__ xl, const float* __restrict__ xr,
                           const float* __restrict__ We, const float* __restrict__ att,
                           float* __restrict__ alpha, unsigned* __restrict__ amax) {
    int gid  = blockIdx.x * blockDim.x + threadIdx.x;
    int wv   = gid >> 5, lane = gid & 31;
    if (wv >= ET * HEADS) return;
    int e = wv >> 1, hd = wv & 1;
    int s, t; float w;
    if (e < N_EDGES) { s = ei[e]; t = ei[N_EDGES + e]; w = ew[e]; }
    else             { s = t = e - N_EDGES; w = ewsum[0] * (1.0f / N_EDGES); }
    const float* xls = xl + (size_t)s * HC + hd * CDIM;
    const float* xrt = xr + (size_t)t * HC + hd * CDIM;
    const float* Weh = We + hd * CDIM;
    const float* ath = att + hd * CDIM;
    float acc = 0.f;
#pragma unroll
    for (int j = 0; j < CDIM / 32; j++) {
        int c = lane + 32 * j;
        float z = lrelu(xls[c] + xrt[c] + w * Weh[c], 0.2f);
        acc = fmaf(z, ath[c], acc);
    }
#pragma unroll
    for (int m = 16; m > 0; m >>= 1) acc += __shfl_xor(acc, m, 32);
    if (lane == 0) {
        alpha[wv] = acc;
        atomicMax(&amax[t * HEADS + hd], enc_ord(acc));
    }
}

// ---------- exp + denom (ex written in place over alpha) ----------
__global__ void edge_ex(const int* __restrict__ ei, float* __restrict__ alpha,
                        const unsigned* __restrict__ amax, float* __restrict__ denom) {
    int w = blockIdx.x * blockDim.x + threadIdx.x;
    if (w >= ET * HEADS) return;
    int e = w >> 1, hd = w & 1;
    int t = (e < N_EDGES) ? ei[N_EDGES + e] : (e - N_EDGES);
    float v = expf(alpha[w] - dec_ord(amax[t * HEADS + hd]));
    alpha[w] = v;
    atomicAdd(&denom[t * HEADS + hd], v);
}

// ---------- weighted aggregation: one wave per (edge, head) ----------
__global__ void edge_accum(const int* __restrict__ ei, const float* __restrict__ xl,
                           const float* __restrict__ ex, const float* __restrict__ denom,
                           float* __restrict__ accum) {
    int gid = blockIdx.x * blockDim.x + threadIdx.x;
    int wv  = gid >> 5, lane = gid & 31;
    if (wv >= ET * HEADS) return;
    int e = wv >> 1, hd = wv & 1;
    int s, t;
    if (e < N_EDGES) { s = ei[e]; t = ei[N_EDGES + e]; }
    else             { s = t = e - N_EDGES; }
    float a = ex[wv] / denom[t * HEADS + hd];
    const float* xls = xl + (size_t)s * HC + hd * CDIM;
    float*       at  = accum + (size_t)t * HC + hd * CDIM;
#pragma unroll
    for (int j = 0; j < CDIM / 32; j++) {
        int c = lane + 32 * j;
        atomicAdd(&at[c], xls[c] * a);
    }
}

// ---------- node finalize: mean-aggr, head-mean, bias, BN, leaky ----------
__global__ void finalize(const float* __restrict__ accum, const float* __restrict__ deg,
                         const float* __restrict__ bias,
                         const float* __restrict__ g, const float* __restrict__ b,
                         const float* __restrict__ m, const float* __restrict__ v,
                         float* __restrict__ xnode, unsigned short* xbf) {
    int t = blockIdx.x * blockDim.x + threadIdx.x;
    if (t >= N_NODES * CDIM) return;
    int n = t / CDIM, c = t % CDIM;
    float d = fmaxf(deg[n], 1.0f);
    float o = (accum[(size_t)n * HC + c] + accum[(size_t)n * HC + CDIM + c]) * (0.5f / d) + bias[c];
    o = (o - m[c]) * rsqrtf(v[c] + 1e-5f) * g[c] + b[c];
    o = lrelu(o, 0.01f);
    xnode[t] = o;
    if (xbf) xbf[t] = f2bf(o);
}

// ---------- weight convert: f32 [128,256] -> bf16 transposed [256,128] ----------
__global__ void wconv_t(const float* __restrict__ W, unsigned short* __restrict__ Wt) {
    int t = blockIdx.x * blockDim.x + threadIdx.x;
    if (t >= CDIM * HC) return;
    int k = t / HC, n = t % HC;
    Wt[(size_t)n * CDIM + k] = f2bf(W[t]);
}

// ---------- layer-2 dual GEMM via bf16 WMMA, async-staged LDS ----------
// One 512-thread block (16 waves) per 16-row stripe of X. Block stages the A
// stripe (16x128 bf16) and BOTH transposed weight matrices (256x128 bf16 each)
// into LDS with GLOBAL_LOAD_ASYNC_TO_LDS_B128, waits on ASYNCcnt, then each
// wave owns one 16-col tile and runs 4 k-steps x 2 weights of
// v_wmma_f32_16x16x32_bf16 fed by ds_load_b128. Rows padded (+16B) so the 16
// row-parallel ds_load_b128's hit staggered banks.
__global__ void __launch_bounds__(512)
gemm_wmma_dual(const unsigned short* __restrict__ xbf,
               const unsigned short* __restrict__ wlt, const unsigned short* __restrict__ wrt,
               const float* __restrict__ bl, const float* __restrict__ br,
               float* __restrict__ xl, float* __restrict__ xr) {
    extern __shared__ unsigned short smem[];
    unsigned short* sA  = smem;                     // 16  x ROWP
    unsigned short* sWl = smem + 16 * ROWP;         // 256 x ROWP
    unsigned short* sWr = sWl + 256 * ROWP;         // 256 x ROWP

    int tid  = threadIdx.x;
    int m0   = blockIdx.x * 16;
    const unsigned short* gA = xbf + (size_t)m0 * CDIM;

    // stage: 16B chunks; A = 256 chunks, each W = 4096 chunks (16 chunks/row)
    for (int c = tid; c < 8448; c += 512) {
        int idx; const unsigned short* gb; unsigned short* lb;
        if (c < 256)       { idx = c;        gb = gA;  lb = sA;  }
        else if (c < 4352) { idx = c - 256;  gb = wlt; lb = sWl; }
        else               { idx = c - 4352; gb = wrt; lb = sWr; }
        int row = idx >> 4, part = idx & 15;
        cp16(gb + row * CDIM + part * 8, lb + row * ROWP + part * 8);
    }
    async_wait0();
    __syncthreads();

    int wave = tid >> 5;         // 0..15 == column tile
    int lane = tid & 31;
    int n0   = wave * 16;
    int half = lane >> 4;
    int l16  = lane & 15;

    const unsigned short* arow  = sA  + (size_t)l16 * ROWP;
    const unsigned short* blrow = sWl + (size_t)(n0 + l16) * ROWP;
    const unsigned short* brrow = sWr + (size_t)(n0 + l16) * ROWP;

    v8f cl = {}; v8f cr = {};
#pragma unroll
    for (int kb = 0; kb < 4; kb++) {
        Frag16 a, bL, bR;
        // A 16x32 k order used here, same as B: lanes0-15 K=0..15, lanes16-31 K=16..31 (contiguous)
        int ak0 = kb * 32 + half * 16;
        a.q[0] = *(const uint4v*)(arow + ak0);
        a.q[1] = *(const uint4v*)(arow + ak0 + 8);
        // B 32x16 k order used here, same as A: lanes0-15 K=0..15, lanes16-31 K=16..31 (contiguous)
        int bk0 = kb * 32 + half * 16;
        bL.q[0] = *(const uint4v*)(blrow + bk0);
        bL.q[1] = *(const uint4v*)(blrow + bk0 + 8);
        bR.q[0] = *(const uint4v*)(brrow + bk0);
        bR.q[1] = *(const uint4v*)(brrow + bk0 + 8);
        cl = __builtin_amdgcn_wmma_f32_16x16x32_bf16(false, a.v, false, bL.v, (short)0, cl, false, false);
        cr = __builtin_amdgcn_wmma_f32_16x16x32_bf16(false, a.v, false, bR.v, (short)0, cr, false, false);
    }
    int n = n0 + l16;
    float bLn = bl[n], bRn = br[n];
#pragma unroll
    for (int r = 0; r < 8; r++) {   // C/D: row = r + 8*half, col = l16
        int m = m0 + r + half * 8;
        xl[(size_t)m * HC + n] = cl[r] + bLn;
        xr[(size_t)m * HC + n] = cr[r] + bRn;
    }
}

// ---------- final FC 128 -> 5 ----------
__global__ void fc_out(const float* __restrict__ x, const float* __restrict__ fcw,
                       const float* __restrict__ fcb, float* __restrict__ out) {
    int t = blockIdx.x * blockDim.x + threadIdx.x;
    if (t >= N_NODES * OUTF) return;
    int n = t / OUTF, o = t % OUTF;
    float acc = fcb[o];
    const float* xr = x + (size_t)n * CDIM;
#pragma unroll 8
    for (int c = 0; c < CDIM; c++) acc = fmaf(xr[c], fcw[c * OUTF + o], acc);
    out[t] = acc;
}

// =====================================================================
extern "C" void kernel_launch(void* const* d_in, const int* in_sizes, int n_in,
                              void* d_out, int out_size, void* d_ws, size_t ws_size,
                              hipStream_t stream) {
    const float* h    = (const float*)d_in[0];
    const int*   ei   = (const int*)d_in[2];
    const float* ew   = (const float*)d_in[3];
    const float* Wl1  = (const float*)d_in[4];  const float* bl1 = (const float*)d_in[5];
    const float* Wr1  = (const float*)d_in[6];  const float* br1 = (const float*)d_in[7];
    const float* We1  = (const float*)d_in[8];  const float* att1 = (const float*)d_in[9];
    const float* bias1= (const float*)d_in[10];
    const float* Wl2  = (const float*)d_in[11]; const float* bl2 = (const float*)d_in[12];
    const float* Wr2  = (const float*)d_in[13]; const float* br2 = (const float*)d_in[14];
    const float* We2  = (const float*)d_in[15]; const float* att2 = (const float*)d_in[16];
    const float* bias2= (const float*)d_in[17];
    const float* bn1g = (const float*)d_in[18]; const float* bn1b = (const float*)d_in[19];
    const float* bn1m = (const float*)d_in[20]; const float* bn1v = (const float*)d_in[21];
    const float* bn2g = (const float*)d_in[22]; const float* bn2b = (const float*)d_in[23];
    const float* bn2m = (const float*)d_in[24]; const float* bn2v = (const float*)d_in[25];
    const float* fcw  = (const float*)d_in[26]; const float* fcb  = (const float*)d_in[27];
    float* out = (float*)d_out;

    // ---- scratch carve ----
    char* wsp = (char*)d_ws;
    size_t off = 0;
    auto take = [&](size_t bytes) -> char* {
        char* p = wsp + off;
        off += (bytes + 255) & ~(size_t)255;
        return p;
    };
    float*          xl    = (float*)take((size_t)N_NODES * HC * 4);
    float*          xr    = (float*)take((size_t)N_NODES * HC * 4);   // reused as accumulator
    float*          alpha = (float*)take((size_t)ET * HEADS * 4);     // becomes ex in place
    unsigned*       amax  = (unsigned*)take((size_t)N_NODES * HEADS * 4);
    float*          denom = (float*)take((size_t)N_NODES * HEADS * 4);
    float*          deg   = (float*)take((size_t)N_NODES * 4);
    float*          xnode = (float*)take((size_t)N_NODES * CDIM * 4);
    unsigned short* xbf   = (unsigned short*)take((size_t)N_NODES * CDIM * 2);
    unsigned short* wlt   = (unsigned short*)take((size_t)CDIM * HC * 2);
    unsigned short* wrt   = (unsigned short*)take((size_t)CDIM * HC * 2);
    float*          ewsm  = (float*)take(4);
    (void)ws_size; (void)in_sizes; (void)n_in; (void)out_size;

    const unsigned NEG_INF_ORD = 0x007FFFFFu;  // enc_ord(-inf)
    const int BS = 256;
    const int nEH = ET * HEADS;                       // 1.8M (edge,head) items
    const int gEHwave = (nEH * 32 + BS - 1) / BS;     // one wave each
    const int gEH = (nEH + BS - 1) / BS;
    const int gNC = (N_NODES * CDIM + BS - 1) / BS;
    const int gNHC = (N_NODES * HC + BS - 1) / BS;
    const size_t GEMM_LDS = (size_t)(16 * ROWP + 2 * 256 * ROWP) * 2;  // ~140 KB of the 320 KB WGP LDS

    // ---- once-per-call init ----
    fill_u32<<<1, 32, 0, stream>>>((unsigned*)ewsm, 0u, 1);
    fill_u32<<<(N_NODES + BS - 1) / BS, BS, 0, stream>>>((unsigned*)deg, 0u, N_NODES);
    ew_sum<<<256, 256, 0, stream>>>(ew, ewsm);
    deg_count<<<(ET + BS - 1) / BS, BS, 0, stream>>>(ei, deg);

    // ================= layer 1 =================
    proj1<<<gNHC, BS, 0, stream>>>(h, Wl1, bl1, Wr1, br1, xl, xr);
    fill_u32<<<(N_NODES * HEADS + BS - 1) / BS, BS, 0, stream>>>(amax, NEG_INF_ORD, N_NODES * HEADS);
    fill_u32<<<(N_NODES * HEADS + BS - 1) / BS, BS, 0, stream>>>((unsigned*)denom, 0u, N_NODES * HEADS);
    edge_alpha<<<gEHwave, BS, 0, stream>>>(ei, ew, ewsm, xl, xr, We1, att1, alpha, amax);
    edge_ex<<<gEH, BS, 0, stream>>>(ei, alpha, amax, denom);
    fill_u32<<<gNHC, BS, 0, stream>>>((unsigned*)xr, 0u, N_NODES * HC);      // xr -> accum
    edge_accum<<<gEHwave, BS, 0, stream>>>(ei, xl, alpha, denom, xr);
    finalize<<<gNC, BS, 0, stream>>>(xr, deg, bias1, bn1g, bn1b, bn1m, bn1v, xnode, xbf);

    // ================= layer 2 =================
    wconv_t<<<(CDIM * HC + BS - 1) / BS, BS, 0, stream>>>(Wl2, wlt);
    wconv_t<<<(CDIM * HC + BS - 1) / BS, BS, 0, stream>>>(Wr2, wrt);
    gemm_wmma_dual<<<N_NODES / 16, 512, GEMM_LDS, stream>>>(xbf, wlt, wrt, bl2, br2, xl, xr);
    fill_u32<<<(N_NODES * HEADS + BS - 1) / BS, BS, 0, stream>>>(amax, NEG_INF_ORD, N_NODES * HEADS);
    fill_u32<<<(N_NODES * HEADS + BS - 1) / BS, BS, 0, stream>>>((unsigned*)denom, 0u, N_NODES * HEADS);
    edge_alpha<<<gEHwave, BS, 0, stream>>>(ei, ew, ewsm, xl, xr, We2, att2, alpha, amax);
    edge_ex<<<gEH, BS, 0, stream>>>(ei, alpha, amax, denom);
    fill_u32<<<gNHC, BS, 0, stream>>>((unsigned*)xr, 0u, N_NODES * HC);
    edge_accum<<<gEHwave, BS, 0, stream>>>(ei, xl, alpha, denom, xr);
    finalize<<<gNC, BS, 0, stream>>>(xr, deg, bias2, bn2g, bn2b, bn2m, bn2v, xnode, nullptr);

    // ================= output head =================
    fc_out<<<(N_NODES * OUTF + BS - 1) / BS, BS, 0, stream>>>(xnode, fcw, fcb, out);
}
